// Net_5669356834646
// MI455X (gfx1250) — compile-verified, matrix operand order checked by reading, not run on hardware
//
#include <hip/hip_runtime.h>
#include <hip/hip_bf16.h>
#include <stddef.h>

// MI455X / gfx1250, wave32. 1x1 convs (77% of FLOPs) run on
// v_wmma_f32_16x16x32_f16; attention gather/scatter on v_fma_mix VALU.
// Activations ping-pong as f16 NHWC in d_ws (2 x 16.78 MB), resident in L2.

typedef _Float16 v8h  __attribute__((ext_vector_type(8)));
typedef _Float16 v16h __attribute__((ext_vector_type(16)));
typedef float    v8f  __attribute__((ext_vector_type(8)));

#define NB   8
#define CH   64
#define HW   128
#define PIX  (HW * HW)          // 16384 pixels per image
#define TS   16                 // spatial tile
#define RDIM 20                 // tile + 2*2 halo
#define RPIX (RDIM * RDIM)      // 400 region pixels = 25 groups of 16

// Load one v16h WMMA operand (K-chunk of 32) from a row of 64 f16 values.
// ISA 16-bit A/B layout: lane half (lane>>4) selects K sub-ranges
//   VGPR0..3: K = half*8 + {0..7};  VGPR4..7: K = 16 + half*8 + {0..7}
__device__ __forceinline__ v16h ld16(const _Float16* p, int hl) {
  union { v16h v; v8h h[2]; } u;
  u.h[0] = *(const v8h*)(p + hl * 8);
  u.h[1] = *(const v8h*)(p + 16 + hl * 8);
  return u.v;
}

// ---------------- conv 3x3, 3->64, ReLU, NCHW f32 -> NHWC f16 --------------
__global__ __launch_bounds__(256) void k_conv_in(const float* __restrict__ x,
                                                 const float* __restrict__ w,
                                                 _Float16* __restrict__ out) {
  __shared__ float sw[64 * 27];          // [o][ci][dy][dx] same flat order as input
  const int tid = threadIdx.x;
  for (int i = tid; i < 64 * 27; i += 256) sw[i] = w[i];
  __syncthreads();

  const int pix = blockIdx.x * 256 + tid;
  const int n = pix >> 14, rem = pix & 16383;
  const int y = rem >> 7, xc = rem & 127;

  float patch[27];
#pragma unroll
  for (int ci = 0; ci < 3; ++ci)
#pragma unroll
    for (int dy = 0; dy < 3; ++dy)
#pragma unroll
      for (int dx = 0; dx < 3; ++dx) {
        const int gy = y + dy - 1, gx = xc + dx - 1;
        patch[ci * 9 + dy * 3 + dx] =
            ((unsigned)gy < 128u && (unsigned)gx < 128u)
                ? x[((n * 3 + ci) << 14) + (gy << 7) + gx] : 0.f;
      }

  _Float16* op = out + (size_t)pix * CH;
  for (int ob = 0; ob < 8; ++ob) {
    v8h hv;
#pragma unroll
    for (int i = 0; i < 8; ++i) {
      const int o = ob * 8 + i;
      float a = 0.f;
#pragma unroll
      for (int j = 0; j < 27; ++j) a += sw[o * 27 + j] * patch[j];
      hv[i] = (_Float16)(a > 0.f ? a : 0.f);
    }
    *(v8h*)(op + ob * 8) = hv;
  }
}

// ---------------- fused local-attention block (WMMA 1x1 convs) -------------
__global__ __launch_bounds__(256) void k_attn(const _Float16* __restrict__ in,
                                              const float* __restrict__ w1,
                                              const float* __restrict__ w2,
                                              const float* __restrict__ w3,
                                              _Float16* __restrict__ out) {
  __shared__ _Float16 s_reg[RPIX * CH];  // 51.2 KB: x1 then x2 then x3 region
  __shared__ _Float16 s_w[CH * CH];      // 8 KB: current weight matrix (f16)

  const int tid  = threadIdx.x;
  const int lane = tid & 31;
  const int wv   = tid >> 5;             // wave id 0..7
  const int col  = lane & 15;            // WMMA row(A)/col(B) = lane%16
  const int hl   = lane >> 4;            // lane half selects K sub-ranges
  const int b    = blockIdx.x;
  const int n    = b >> 6;
  const int t    = b & 63;
  const int ty0  = (t >> 3) << 4;
  const int tx0  = (t & 7) << 4;

  v16h A[4][2];                          // 4 M-tiles x 2 K-chunks of weights

  auto load_w = [&](const float* w) {
    for (int i = tid; i < CH * CH; i += 256) s_w[i] = (_Float16)w[i];
  };
  auto load_A = [&]() {
#pragma unroll
    for (int m = 0; m < 4; ++m) {
      const _Float16* rp = s_w + (m * 16 + col) * CH;
      A[m][0] = ld16(rp, hl);
      A[m][1] = ld16(rp + 32, hl);
    }
  };
  // One 64ch x 16pixel GEMM group: Y = W(64x64) * X(64x16), f16 in, f32 acc.
  auto gemm_store = [&](int r, int gy, int gx) {
    v16h b0 = {}, b1 = {};
    if ((unsigned)gy < 128u && (unsigned)gx < 128u) {   // zero-padded unfold
      const _Float16* gp = in + (size_t)(((n << 7) + gy) * 128 + gx) * CH;
      b0 = ld16(gp, hl);
      b1 = ld16(gp + 32, hl);
    }
#pragma unroll
    for (int m = 0; m < 4; ++m) {
      v8f acc = {};
      acc = __builtin_amdgcn_wmma_f32_16x16x32_f16(false, A[m][0], false, b0,
                                                   (short)0, acc, false, false);
      acc = __builtin_amdgcn_wmma_f32_16x16x32_f16(false, A[m][1], false, b1,
                                                   (short)0, acc, false, false);
      v8h hv;                            // D vgpr v -> channel m*16 + 8*hl + v
#pragma unroll
      for (int i = 0; i < 8; ++i) hv[i] = (_Float16)acc[i];
      *(v8h*)(s_reg + r * CH + m * 16 + hl * 8) = hv;   // ds_store_b128
    }
  };

  // ---- x1 = W1 * h on inner 16x16 ----
  load_w(w1);
  __syncthreads();
  load_A();
  for (int g = wv; g < 16; g += 8)
    gemm_store(g * 16 + col, ty0 + g, tx0 + col);
  __syncthreads();

  // per-thread pixel: pull x1[:,p] into registers
  const int iy = tid >> 4, ix = tid & 15;
  float x1f[CH];
#pragma unroll
  for (int c = 0; c < CH; ++c) x1f[c] = (float)s_reg[tid * CH + c];
  __syncthreads();

  // ---- x2 = W2 * h over 20x20 halo region ----
  load_w(w2);
  __syncthreads();
  load_A();
  for (int g = wv; g < 25; g += 8) {
    const int r = g * 16 + col;
    gemm_store(r, ty0 + (r / RDIM) - 2, tx0 + (r % RDIM) - 2);
  }
  __syncthreads();

  // similarity (25 x 64 MACs/pixel) + softmax, all in registers
  float sim[25];
#pragma unroll
  for (int dy = 0; dy < 5; ++dy)
#pragma unroll
    for (int dx = 0; dx < 5; ++dx) {
      const _Float16* q = s_reg + ((iy + dy) * RDIM + ix + dx) * CH;
      float a = 0.f;
#pragma unroll
      for (int c = 0; c < CH; ++c) a += x1f[c] * (float)q[c];
      sim[dy * 5 + dx] = a;
    }
  float mx = sim[0];
#pragma unroll
  for (int k = 1; k < 25; ++k) mx = fmaxf(mx, sim[k]);
  float ssum = 0.f;
#pragma unroll
  for (int k = 0; k < 25; ++k) { sim[k] = __expf(sim[k] - mx); ssum += sim[k]; }
  const float inv = 1.0f / ssum;
#pragma unroll
  for (int k = 0; k < 25; ++k) sim[k] *= inv;
  __syncthreads();

  // ---- x3 = W3 * h over region (reuse LDS) ----
  load_w(w3);
  __syncthreads();
  load_A();
  for (int g = wv; g < 25; g += 8) {
    const int r = g * 16 + col;
    gemm_store(r, ty0 + (r / RDIM) - 2, tx0 + (r % RDIM) - 2);
  }
  __syncthreads();

  // weighted gather over 25 offsets, ReLU, store NHWC f16
  _Float16* op =
      out + (size_t)(((n << 7) + ty0 + iy) * 128 + tx0 + ix) * CH;
  for (int cb = 0; cb < 8; ++cb) {
    float acc[8] = {0, 0, 0, 0, 0, 0, 0, 0};
#pragma unroll
    for (int dy = 0; dy < 5; ++dy)
#pragma unroll
      for (int dx = 0; dx < 5; ++dx) {
        const float wk = sim[dy * 5 + dx];
        const v8h vv =
            *(const v8h*)(s_reg + ((iy + dy) * RDIM + ix + dx) * CH + cb * 8);
#pragma unroll
        for (int i = 0; i < 8; ++i) acc[i] += wk * (float)vv[i];
      }
    v8h hv;
#pragma unroll
    for (int i = 0; i < 8; ++i) {
      const float a = acc[i];
      hv[i] = (_Float16)(a > 0.f ? a : 0.f);
    }
    *(v8h*)(op + cb * 8) = hv;
  }
}

// ---------------- conv 3x3, 64->3, + residual, f32 out ---------------------
__global__ __launch_bounds__(256) void k_conv_out(const _Float16* __restrict__ h,
                                                  const float* __restrict__ w,
                                                  const float* __restrict__ x,
                                                  float* __restrict__ out) {
  __shared__ float sw[3 * 9 * 64];       // remapped [o][tap][c] for unit-stride c
  const int tid = threadIdx.x;
  for (int i = tid; i < 1728; i += 256) {
    const int o = i / 576, rem = i % 576, c = rem / 9, tp = rem % 9;
    sw[(o * 9 + tp) * 64 + c] = w[i];
  }
  __syncthreads();

  const int pix = blockIdx.x * 256 + tid;
  const int n = pix >> 14, rem = pix & 16383;
  const int y = rem >> 7, xc = rem & 127;

  float a0 = 0.f, a1 = 0.f, a2 = 0.f;
  for (int dy = 0; dy < 3; ++dy)
    for (int dx = 0; dx < 3; ++dx) {
      const int gy = y + dy - 1, gx = xc + dx - 1;
      if ((unsigned)gy < 128u && (unsigned)gx < 128u) {
        const _Float16* q = h + (size_t)((n << 14) + (gy << 7) + gx) * 64;
        const int tp = dy * 3 + dx;
        const float* w0 = sw + (0 * 9 + tp) * 64;
        const float* w1 = sw + (1 * 9 + tp) * 64;
        const float* w2 = sw + (2 * 9 + tp) * 64;
#pragma unroll
        for (int c = 0; c < 64; ++c) {
          const float v = (float)q[c];
          a0 += w0[c] * v;
          a1 += w1[c] * v;
          a2 += w2[c] * v;
        }
      }
    }
  const int sp = (y << 7) + xc;
  const int base = (n * 3) << 14;
  out[base + sp]             = a0 + x[base + sp];
  out[base + 16384 + sp]     = a1 + x[base + 16384 + sp];
  out[base + 2 * 16384 + sp] = a2 + x[base + 2 * 16384 + sp];
}

extern "C" void kernel_launch(void* const* d_in, const int* in_sizes, int n_in,
                              void* d_out, int out_size, void* d_ws, size_t ws_size,
                              hipStream_t stream) {
  (void)in_sizes; (void)n_in; (void)out_size; (void)ws_size;
  const float* x    = (const float*)d_in[0];
  const float* wci  = (const float*)d_in[1];
  const float* w1_0 = (const float*)d_in[2];
  const float* w2_0 = (const float*)d_in[3];
  const float* w3_0 = (const float*)d_in[4];
  const float* w1_1 = (const float*)d_in[5];
  const float* w2_1 = (const float*)d_in[6];
  const float* w3_1 = (const float*)d_in[7];
  const float* w1_2 = (const float*)d_in[8];
  const float* w2_2 = (const float*)d_in[9];
  const float* w3_2 = (const float*)d_in[10];
  const float* wco  = (const float*)d_in[11];
  float* out = (float*)d_out;

  // two NHWC f16 activation buffers (16.78 MB each) in workspace
  char* ws = (char*)d_ws;
  _Float16* bufA = (_Float16*)ws;
  _Float16* bufB = (_Float16*)(ws + (size_t)NB * PIX * CH * sizeof(_Float16));

  const dim3 blk(256);
  const int grid_px   = (NB * PIX) / 256;      // 512
  const int grid_attn = NB * (HW / TS) * (HW / TS);  // 512

  k_conv_in <<<grid_px,   blk, 0, stream>>>(x, wci, bufA);
  k_attn    <<<grid_attn, blk, 0, stream>>>(bufA, w1_0, w2_0, w3_0, bufB);
  k_attn    <<<grid_attn, blk, 0, stream>>>(bufB, w1_1, w2_1, w3_1, bufA);
  k_attn    <<<grid_attn, blk, 0, stream>>>(bufA, w1_2, w2_2, w3_2, bufB);
  k_conv_out<<<grid_px,   blk, 0, stream>>>(bufB, wco, x, out);
}
